// ModelNet40xConv_10505490006260
// MI455X (gfx1250) — matrix operand order repaired and checked by reading, not compiled, not run on hardware
//
#include <hip/hip_runtime.h>
#include <math.h>

typedef __attribute__((ext_vector_type(16))) _Float16 v16h;
typedef __attribute__((ext_vector_type(8)))  float    v8f;

// --- CDNA5 async global->LDS copy (ASYNCcnt-tracked), ISA §15.18.3 op 96 ---
__device__ __forceinline__ void async_copy_b32(unsigned lds_off, const float* gptr)
{
    unsigned long long ga = (unsigned long long)(uintptr_t)gptr;
    asm volatile("global_load_async_to_lds_b32 %0, %1, off"
                 :: "v"(lds_off), "v"(ga) : "memory");
}
__device__ __forceinline__ void wait_async()
{
    asm volatile("s_wait_asynccnt 0" ::: "memory");
}

// ---------------------------------------------------------------------------
// WMMA GEMM: C[M,N] = act(A[M,K] * B (+bias)), f16 operands, f32 accumulate.
// A row-major lda. BT==0 -> B is (K,N) ldb ; BT==1 -> B is (N,K) ldb.
// BT is a template parameter so the B-addressing mode resolves at compile
// time (no per-element scalar branches in the hot loop). 4 waves per block,
// one 16x16 C tile per wave. Row/col OOB lanes load a clamped (valid) address
// and compute garbage that is never stored -- only the ragged K tail is
// zero-padded. EXEC stays all-ones through both WMMAs. act: 0 none, 1 ELU.
// ---------------------------------------------------------------------------
template <int BT>
__launch_bounds__(128)
__global__ void gemm_wmma_kernel(const float* __restrict__ A, int lda,
                                 const float* __restrict__ Bm, int ldb,
                                 const float* __restrict__ bias,
                                 float* __restrict__ C, int ldc,
                                 int M, int N, int Kd, int act)
{
    const int lane  = threadIdx.x & 31;
    const int wave  = threadIdx.x >> 5;
    const int tileM = (blockIdx.x * 4 + wave) * 16;
    const int tileN = blockIdx.y * 16;

    const int m  = lane & 15;
    int gr = tileM + m; if (gr > M - 1) gr = M - 1;   // clamp: address safety only
    int gn = tileN + m; if (gn > N - 1) gn = N - 1;
    const float* Arow = A + (size_t)gr * lda;
    const float* Bbase = BT ? (Bm + (size_t)gn * ldb) : (Bm + gn);

    const int abase = (lane < 16) ? 0 : 16;           // A: contiguous K split
    const int bbase = (lane < 16) ? 0 : 16;           // B: contiguous K split

    v8f acc = {};
    int k0 = 0;
    for (; k0 + 32 <= Kd; k0 += 32) {
        if (k0 + 32 < Kd) {                           // emits global_prefetch_b8
            __builtin_prefetch(Arow + k0 + 32, 0, 3);
            __builtin_prefetch(BT ? (Bbase + k0 + 32) : (Bbase + (size_t)(k0 + 32) * ldb), 0, 3);
        }
        union { v16h v; _Float16 h[16]; } ua, ub;
        for (int t = 0; t < 8; ++t) {
            int ka = k0 + abase + 2 * t;
            ua.h[2 * t]     = (_Float16)Arow[ka];
            ua.h[2 * t + 1] = (_Float16)Arow[ka + 1];
            int kb = k0 + bbase + 2 * t;
            float b0 = BT ? Bbase[kb]     : Bbase[(size_t)kb * ldb];
            float b1 = BT ? Bbase[kb + 1] : Bbase[(size_t)(kb + 1) * ldb];
            ub.h[2 * t]     = (_Float16)b0;
            ub.h[2 * t + 1] = (_Float16)b1;
        }
        acc = __builtin_amdgcn_wmma_f32_16x16x32_f16(
                  false, ua.v, false, ub.v, (short)0, acc, false, false);
    }
    if (k0 < Kd) {                                    // ragged K tail: zero-pad
        const int Kmax = Kd - 1;
        union { v16h v; _Float16 h[16]; } ua, ub;
        for (int t = 0; t < 8; ++t) {
            int ka = k0 + abase + 2 * t;
            int c0 = (ka     < Kmax) ? ka     : Kmax;
            int c1 = (ka + 1 < Kmax) ? ka + 1 : Kmax;
            float a0 = Arow[c0];
            float a1 = Arow[c1];
            ua.h[2 * t]     = (_Float16)((ka     < Kd) ? a0 : 0.0f);
            ua.h[2 * t + 1] = (_Float16)((ka + 1 < Kd) ? a1 : 0.0f);
            int kb = k0 + bbase + 2 * t;
            int d0 = (kb     < Kmax) ? kb     : Kmax;
            int d1 = (kb + 1 < Kmax) ? kb + 1 : Kmax;
            float b0 = BT ? Bbase[d0] : Bbase[(size_t)d0 * ldb];
            float b1 = BT ? Bbase[d1] : Bbase[(size_t)d1 * ldb];
            ub.h[2 * t]     = (_Float16)((kb     < Kd) ? b0 : 0.0f);
            ub.h[2 * t + 1] = (_Float16)((kb + 1 < Kd) ? b1 : 0.0f);
        }
        acc = __builtin_amdgcn_wmma_f32_16x16x32_f16(
                  false, ua.v, false, ub.v, (short)0, acc, false, false);
    }

    // Epilogue: C layout m = vgpr + 8*(lane>=16), n = lane&15.
    union { v8f v; float f[8]; } uc; uc.v = acc;
    int nn = tileN + (lane & 15);
    int mb = tileM + (lane >> 4) * 8;
    if (nn < N) {
        float bv = bias ? bias[nn] : 0.0f;
        for (int r = 0; r < 8; ++r) {
            int mm = mb + r;
            if (mm < M) {
                float v = uc.f[r] + bv;
                if (act == 1) v = (v > 0.0f) ? v : expm1f(v);
                C[(size_t)mm * ldc + nn] = v;
            }
        }
    }
}

// ---------------------------------------------------------------------------
// KNN: per point, top K*dil nearest (d2 ascending, lower index wins ties =
// jax.lax.top_k semantics), then take every dil-th. One block per
// (batch, 128-point chunk); the batch's point cloud (n <= 1024 -> 12 KB) is
// staged into LDS with async global->LDS loads and scanned from there.
// ---------------------------------------------------------------------------
__global__ void knn_kernel(const float* __restrict__ pos, int n,
                           int K, int dil, int* __restrict__ idx)
{
    __shared__ float sp[1024 * 3];
    int b = blockIdx.y;
    const float* pb = pos + (size_t)b * n * 3;
    for (int j = threadIdx.x; j < n * 3; j += blockDim.x)
        async_copy_b32((unsigned)(uintptr_t)&sp[j], pb + j);
    wait_async();
    __syncthreads();

    int i = blockIdx.x * blockDim.x + threadIdx.x;
    if (i >= n) return;
    float xi = sp[i * 3], yi = sp[i * 3 + 1], zi = sp[i * 3 + 2];
    int KD = K * dil;                       // <= 32
    float bd[32]; int bi[32];
    int cnt = 0;
    for (int j = 0; j < n; ++j) {
        float dx = sp[j * 3] - xi, dy = sp[j * 3 + 1] - yi, dz = sp[j * 3 + 2] - zi;
        float d = dx * dx + dy * dy + dz * dz;
        if (cnt == KD && d >= bd[KD - 1]) continue;
        int q = (cnt < KD) ? cnt : KD - 1;
        while (q > 0 && bd[q - 1] > d) { bd[q] = bd[q - 1]; bi[q] = bi[q - 1]; --q; }
        bd[q] = d; bi[q] = j;
        if (cnt < KD) ++cnt;
    }
    size_t p = (size_t)b * n + i;
    for (int k = 0; k < K; ++k)
        idx[p * K + k] = bi[k * dil];
}

// rel[p,k,:] = pos[nb]-pos[i]; xs[p,k,Cd..Cd+Cin) = x[nb]. Thread per (p,k).
__global__ void gather_rel_x_kernel(const float* __restrict__ pos,
                                    const float* __restrict__ x,
                                    const int* __restrict__ idx,
                                    int B, int n, int K, int Cin, int Cd, int C,
                                    float* __restrict__ rel, float* __restrict__ xs)
{
    int t = blockIdx.x * blockDim.x + threadIdx.x;
    if (t >= B * n * K) return;
    int p = t / K; int b = p / n; int i = p % n;
    int j = idx[t];
    const float* pb = pos + (size_t)b * n * 3;
    rel[(size_t)t * 3 + 0] = pb[j * 3 + 0] - pb[i * 3 + 0];
    rel[(size_t)t * 3 + 1] = pb[j * 3 + 1] - pb[i * 3 + 1];
    rel[(size_t)t * 3 + 2] = pb[j * 3 + 2] - pb[i * 3 + 2];
    const float* xr = x + ((size_t)b * n + j) * Cin;
    float* xw = xs + (size_t)t * C + Cd;
    for (int c = 0; c < Cin; ++c) xw[c] = xr[c];
}

// ---------------------------------------------------------------------------
// Farthest point sampling. One block per batch; sequential argmax loop,
// first-max tie-break (jnp.argmax). mind and the point cloud live in LDS;
// the point cloud is staged with async global->LDS loads.
// ---------------------------------------------------------------------------
__global__ void fps_kernel(const float* __restrict__ pos, int n, int m,
                           int* __restrict__ out)
{
    __shared__ float sp[1024 * 3];
    __shared__ float mind[1024];
    __shared__ float rv[256];
    __shared__ int   ri[256];
    int b = blockIdx.x;
    const float* pb = pos + (size_t)b * n * 3;
    int tid = threadIdx.x, nt = blockDim.x;
    for (int j = tid; j < n * 3; j += nt)
        async_copy_b32((unsigned)(uintptr_t)&sp[j], pb + j);
    wait_async();
    __syncthreads();

    float x0 = sp[0], y0 = sp[1], z0 = sp[2];
    for (int j = tid; j < n; j += nt) {
        float dx = sp[j * 3] - x0, dy = sp[j * 3 + 1] - y0, dz = sp[j * 3 + 2] - z0;
        mind[j] = dx * dx + dy * dy + dz * dz;
    }
    if (tid == 0) out[(size_t)b * m] = 0;
    __syncthreads();
    for (int it = 1; it < m; ++it) {
        float bv = -1.0f; int bi = 0x7fffffff;
        for (int j = tid; j < n; j += nt) {
            float v = mind[j];
            if (v > bv) { bv = v; bi = j; }
        }
        rv[tid] = bv; ri[tid] = bi;
        __syncthreads();
        for (int s = nt / 2; s > 0; s >>= 1) {
            if (tid < s) {
                if (rv[tid + s] > rv[tid] ||
                    (rv[tid + s] == rv[tid] && ri[tid + s] < ri[tid])) {
                    rv[tid] = rv[tid + s]; ri[tid] = ri[tid + s];
                }
            }
            __syncthreads();
        }
        int nxt = ri[0];
        if (tid == 0) out[(size_t)b * m + it] = nxt;
        float xn = sp[nxt * 3], yn = sp[nxt * 3 + 1], zn = sp[nxt * 3 + 2];
        for (int j = tid; j < n; j += nt) {
            float dx = sp[j * 3] - xn, dy = sp[j * 3 + 1] - yn, dz = sp[j * 3 + 2] - zn;
            float d = dx * dx + dy * dy + dz * dz;
            if (d < mind[j]) mind[j] = d;
        }
        __syncthreads();
    }
}

__global__ void gather_points_kernel(const float* __restrict__ x,
                                     const float* __restrict__ pos,
                                     const int* __restrict__ idx,
                                     int B, int n, int m, int C,
                                     float* __restrict__ xo, float* __restrict__ po)
{
    int t = blockIdx.x * blockDim.x + threadIdx.x;
    if (t >= B * m) return;
    int b = t / m;
    int j = idx[t];
    const float* xr = x + ((size_t)b * n + j) * C;
    float* xw = xo + (size_t)t * C;
    for (int c = 0; c < C; ++c) xw[c] = xr[c];
    const float* pr = pos + ((size_t)b * n + j) * 3;
    float* pw = po + (size_t)t * 3;
    pw[0] = pr[0]; pw[1] = pr[1]; pw[2] = pr[2];
}

// ---------------------------------------------------------------------------
// BatchNorm: two-pass (stats per channel, then apply). ld = row stride so we
// can normalize a column-slice of a wider buffer. biased variance (jnp.var).
// ---------------------------------------------------------------------------
__global__ void bn_stats_kernel(const float* __restrict__ x, int M, int ld,
                                float* __restrict__ mean, float* __restrict__ rstd)
{
    __shared__ float ss[256], sq[256];
    int c = blockIdx.x;
    float s = 0.0f, q = 0.0f;
    for (int r = threadIdx.x; r < M; r += blockDim.x) {
        float v = x[(size_t)r * ld + c];
        s += v; q += v * v;
    }
    ss[threadIdx.x] = s; sq[threadIdx.x] = q;
    __syncthreads();
    for (int st = blockDim.x / 2; st > 0; st >>= 1) {
        if (threadIdx.x < st) {
            ss[threadIdx.x] += ss[threadIdx.x + st];
            sq[threadIdx.x] += sq[threadIdx.x + st];
        }
        __syncthreads();
    }
    if (threadIdx.x == 0) {
        float mu = ss[0] / (float)M;
        float var = sq[0] / (float)M - mu * mu;
        mean[c] = mu;
        rstd[c] = rsqrtf(var + 1e-5f);
    }
}

__global__ void bn_apply_kernel(float* __restrict__ x, int M, int C, int ld,
                                const float* __restrict__ mean,
                                const float* __restrict__ rstd,
                                const float* __restrict__ g,
                                const float* __restrict__ be, int relu)
{
    int t = blockIdx.x * blockDim.x + threadIdx.x;
    if (t >= M * C) return;
    int r = t / C, c = t % C;
    float v = (x[(size_t)r * ld + c] - mean[c]) * rstd[c] * g[c] + be[c];
    if (relu) v = fmaxf(v, 0.0f);
    x[(size_t)r * ld + c] = v;
}

// y[p,c] = bf[c] + sum_k xs[p,k,c] * (sum_j t[p,k,j]*Wf[c,j])
__global__ void xt_combine_kernel(const float* __restrict__ xs,
                                  const float* __restrict__ t,
                                  const float* __restrict__ Wf,
                                  const float* __restrict__ bf,
                                  int P, int K, int C, float* __restrict__ y)
{
    int tid = blockIdx.x * blockDim.x + threadIdx.x;
    if (tid >= P * C) return;
    int c = tid % C, p = tid / C;
    const float* xsp = xs + (size_t)p * K * C;
    const float* tp  = t + (size_t)p * K * K;
    float acc = bf[c];
    for (int k = 0; k < K; ++k) {
        float s = 0.0f;
        for (int j = 0; j < K; ++j) s += tp[k * K + j] * Wf[c * K + j];
        acc += xsp[(size_t)k * C + c] * s;
    }
    y[(size_t)p * C + c] = acc;
}

__global__ void mean_kernel(const float* __restrict__ x, int B, int n, int C,
                            float* __restrict__ out)
{
    int t = blockIdx.x * blockDim.x + threadIdx.x;
    if (t >= B * C) return;
    int b = t / C, c = t % C;
    float s = 0.0f;
    for (int i = 0; i < n; ++i) s += x[((size_t)b * n + i) * C + c];
    out[t] = s / (float)n;
}

// ===========================================================================
// Host orchestration
// ===========================================================================
struct Cv {
    const float *W1a,*b1a,*g1a,*be1a,*W1b,*b1b,*g1b,*be1b,
                *W2a,*b2a,*g2a,*be2a,*Wc1,*bc1,*g2b,*be2b,
                *Wc2,*bc2,*g2c,*be2c,*Wf,*bf,*Wl,*bl;
};

static inline unsigned cdiv(unsigned a, unsigned b) { return (a + b - 1) / b; }

static inline void launch_gemm(const float* A, int lda, const float* B, int ldb,
                               int bT, const float* bias, float* C, int ldc,
                               int M, int N, int K, int act, hipStream_t s)
{
    dim3 g(cdiv(M, 64), cdiv(N, 16));
    if (bT)
        gemm_wmma_kernel<1><<<g, 128, 0, s>>>(A, lda, B, ldb, bias, C, ldc, M, N, K, act);
    else
        gemm_wmma_kernel<0><<<g, 128, 0, s>>>(A, lda, B, ldb, bias, C, ldc, M, N, K, act);
}

struct Arena {
    float *rel, *h, *xs, *ta, *tb, *y;
    float *mean, *rstd;
    int *idxK;
};

static inline void run_bn(float* x, int M, int C, int ld, const float* g,
                          const float* be, int relu, Arena& A, hipStream_t s)
{
    bn_stats_kernel<<<C, 256, 0, s>>>(x, M, ld, A.mean, A.rstd);
    bn_apply_kernel<<<cdiv(M * C, 256), 256, 0, s>>>(x, M, C, ld, A.mean, A.rstd, g, be, relu);
}

static void run_xconv(const float* x, const float* pos, const Cv& P,
                      int B, int n, int K, int dil, int Cin, int Cd, int Cout,
                      Arena& A, float* out, hipStream_t s)
{
    int C  = Cin + Cd;
    int Pn = B * n;
    // KNN + gathers
    {
        dim3 g(cdiv(n, 128), B);
        knn_kernel<<<g, 128, 0, s>>>(pos, n, K, dil, A.idxK);
    }
    gather_rel_x_kernel<<<cdiv(Pn * K, 256), 256, 0, s>>>(pos, x, A.idxK, B, n, K,
                                                          Cin, Cd, C, A.rel, A.xs);
    // h = bn(elu(rel @ W1a + b1a))
    launch_gemm(A.rel, 3, P.W1a, Cd, 0, P.b1a, A.h, Cd, Pn * K, Cd, 3, 1, s);
    run_bn(A.h, Pn * K, Cd, Cd, P.g1a, P.be1a, 0, A, s);
    // h = bn(elu(h @ W1b + b1b)) -> xs[:, 0:Cd]
    launch_gemm(A.h, Cd, P.W1b, Cd, 0, P.b1b, A.xs, C, Pn * K, Cd, Cd, 1, s);
    run_bn(A.xs, Pn * K, Cd, C, P.g1b, P.be1b, 0, A, s);
    // t = bn(elu(rel_flat @ W2a + b2a))           (Pn, K*3) -> (Pn, K*K)
    launch_gemm(A.rel, K * 3, P.W2a, K * K, 0, P.b2a, A.ta, K * K, Pn, K * K, K * 3, 1, s);
    run_bn(A.ta, Pn, K * K, K * K, P.g2a, P.be2a, 0, A, s);
    // tc1: per g, (Pn,K) @ Wc1[g]^T + bc1[g], elu, then bn
    for (int g = 0; g < K; ++g)
        launch_gemm(A.ta + g * K, K * K, P.Wc1 + (size_t)g * K * K, K, 1,
                    P.bc1 + g * K, A.tb + g * K, K * K, Pn, K, K, 1, s);
    run_bn(A.tb, Pn, K * K, K * K, P.g2b, P.be2b, 0, A, s);
    // tc2: per g, (Pn,K) @ Wc2[g]^T + bc2[g], then bn
    for (int g = 0; g < K; ++g)
        launch_gemm(A.tb + g * K, K * K, P.Wc2 + (size_t)g * K * K, K, 1,
                    P.bc2 + g * K, A.ta + g * K, K * K, Pn, K, K, 0, s);
    run_bn(A.ta, Pn, K * K, K * K, P.g2c, P.be2c, 0, A, s);
    // y = einsum(xs,t) contracted with Wf + bf
    xt_combine_kernel<<<cdiv(Pn * C, 256), 256, 0, s>>>(A.xs, A.ta, P.Wf, P.bf, Pn, K, C, A.y);
    // out = y @ Wl + bl
    launch_gemm(A.y, C, P.Wl, Cout, 0, P.bl, out, Cout, Pn, Cout, C, 0, s);
}

static Cv get_cv(void* const* d_in, int base)
{
    Cv c;
    const float** p = (const float**)&c;
    for (int i = 0; i < 24; ++i) p[i] = (const float*)d_in[base + i];
    return c;
}

extern "C" void kernel_launch(void* const* d_in, const int* in_sizes, int n_in,
                              void* d_out, int out_size, void* d_ws, size_t ws_size,
                              hipStream_t stream)
{
    (void)in_sizes; (void)n_in; (void)out_size; (void)ws_size;
    const float* x0   = (const float*)d_in[0];
    const float* pos0 = (const float*)d_in[1];
    Cv cv1 = get_cv(d_in, 2);
    Cv cv2 = get_cv(d_in, 26);
    Cv cv3 = get_cv(d_in, 50);
    Cv cv4 = get_cv(d_in, 74);
    Cv cv5 = get_cv(d_in, 98);
    const float *bn1g = (const float*)d_in[122], *bn1b = (const float*)d_in[123];
    const float *bn2g = (const float*)d_in[124], *bn2b = (const float*)d_in[125];
    const float *bn3g = (const float*)d_in[126], *bn3b = (const float*)d_in[127];
    const float *bn4g = (const float*)d_in[128], *bn4b = (const float*)d_in[129];
    const float *bn5g = (const float*)d_in[130], *bn5b = (const float*)d_in[131];
    const float *l1W = (const float*)d_in[132], *l1b = (const float*)d_in[133];
    const float *l2W = (const float*)d_in[134], *l2b = (const float*)d_in[135];

    // workspace arena (floats)
    float* f = (float*)d_ws;
    size_t o = 0;
    Arena A;
    A.rel = f + o; o +=  786432;   // B*n*K*3        max (L1)
    A.h   = f + o; o += 8388608;   // B*n*K*Cd       max (L1)
    A.xs  = f + o; o += 9175040;   // B*n*K*(Cd+Cin) max (L1)
    A.ta  = f + o; o += 2097152;   // B*n*K*K        max (L1)
    A.tb  = f + o; o += 2097152;
    A.y   = f + o; o += 1146880;   // B*n*C          max (L1)
    float* xA   = f + o; o += 1048576;  // inter-layer features
    float* xB   = f + o; o += 1048576;
    float* posB = f + o; o +=   98304;
    float* posC = f + o; o +=   98304;
    A.mean = f + o; o += 1024;
    A.rstd = f + o; o += 1024;
    float* feat = f + o; o += 8192;     // (32,256)
    float* tmp  = f + o; o += 2048;     // (32,64)
    A.idxK = (int*)(f + o); o += 262144;
    int* idxF  = (int*)(f + o); o += 16384;

    const int B = 32;
    hipStream_t s = stream;

    // ---- layer 1: n=1024, K=8, dil=1, 3 -> (32d) -> 32
    run_xconv(x0, pos0, cv1, B, 1024, 8, 1, 3, 32, 32, A, xA, s);
    fps_kernel<<<B, 256, 0, s>>>(pos0, 1024, 338, idxF);
    gather_points_kernel<<<cdiv(B * 338, 256), 256, 0, s>>>(xA, pos0, idxF, B, 1024, 338, 32, xB, posB);
    run_bn(xB, B * 338, 32, 32, bn1g, bn1b, 1, A, s);

    // ---- layer 2: n=338, K=8, dil=2, 32 -> (64d) -> 64
    run_xconv(xB, posB, cv2, B, 338, 8, 2, 32, 64, 64, A, xA, s);
    run_bn(xA, B * 338, 64, 64, bn2g, bn2b, 1, A, s);
    fps_kernel<<<B, 256, 0, s>>>(posB, 338, 112, idxF);
    gather_points_kernel<<<cdiv(B * 112, 256), 256, 0, s>>>(xA, posB, idxF, B, 338, 112, 64, xB, posC);

    // ---- layer 3: n=112, K=12, dil=2, 64 -> (128d) -> 128
    run_xconv(xB, posC, cv3, B, 112, 12, 2, 64, 128, 128, A, xA, s);
    run_bn(xA, B * 112, 128, 128, bn3g, bn3b, 1, A, s);
    fps_kernel<<<B, 256, 0, s>>>(posC, 112, 37, idxF);
    gather_points_kernel<<<cdiv(B * 37, 256), 256, 0, s>>>(xA, posC, idxF, B, 112, 37, 128, xB, posB);

    // ---- layer 4: n=37, K=16, dil=2, 128 -> (128d) -> 128
    run_xconv(xB, posB, cv4, B, 37, 16, 2, 128, 128, 128, A, xA, s);
    run_bn(xA, B * 37, 128, 128, bn4g, bn4b, 1, A, s);

    // ---- layer 5: n=37, K=16, dil=2, 128 -> (256d) -> 256
    run_xconv(xA, posB, cv5, B, 37, 16, 2, 128, 256, 256, A, xB, s);

    // ---- head: mean over points, bn5+relu, lin1, lin2
    mean_kernel<<<cdiv(B * 256, 256), 256, 0, s>>>(xB, B, 37, 256, feat);
    run_bn(feat, B, 256, 256, bn5g, bn5b, 1, A, s);
    launch_gemm(feat, 256, l1W, 64, 0, l1b, tmp, 64, B, 64, 256, 0, s);
    launch_gemm(tmp, 64, l2W, 40, 0, l2b, (float*)d_out, 40, B, 40, 64, 0, s);
}
